// PINN_67087389164119
// MI455X (gfx1250) — matrix operand order repaired and checked by reading, not compiled, not run on hardware
//
#include <hip/hip_runtime.h>
#include <hip/hip_bf16.h>
#include <math.h>

typedef __attribute__((ext_vector_type(16))) __bf16 bf16x16;
typedef __attribute__((ext_vector_type(8)))  float  f32x8;

#define HIDDEN 128
#define TILE_M 16

union AFrag { bf16x16 v; uint4 q[2]; };

// --------------------------------------------------------------------------
// Pack W2/W3/W4 (128x128 f32 row-major) into bf16 WMMA B-fragment layout:
//   pk[ ((L*4 + kb)*8 + nt)*32*16 + lane*16 + e ]
//   lane 0-15 : K = kb*32 + e      (e = 0..15), N = nt*16 + lane
//   lane 16-31: K = kb*32 + 16 + e,             N = nt*16 + (lane-16)
// so each lane's 16 bf16 (32B) is one contiguous load in the main kernel.
// --------------------------------------------------------------------------
__global__ void pack_weights_kernel(const float* __restrict__ W2,
                                    const float* __restrict__ W3,
                                    const float* __restrict__ W4,
                                    __bf16* __restrict__ pk) {
  int idx = blockIdx.x * blockDim.x + threadIdx.x;
  if (idx >= 3 * 4 * 8 * 32 * 16) return;
  int e    = idx & 15;
  int lane = (idx >> 4) & 31;
  int nt   = (idx >> 9) & 7;
  int kb   = (idx >> 12) & 3;
  int L    = idx >> 14;
  const float* W = (L == 0) ? W2 : ((L == 1) ? W3 : W4);
  int K = kb * 32 + ((lane >= 16) ? 16 : 0) + e;
  int N = nt * 16 + (lane & 15);
  pk[idx] = (__bf16)W[K * HIDDEN + N];
}

// NST: 1 (values), 3 (+Jacobian), 5 (+d2/dx2, d2/dy2). State index map:
//   0=h, 1=h_x, 2=h_y, 3=h_xx, 4=h_yy
// MODE: 0 -> (u, v, u)   1 -> (p, p, p)   2 -> (u, v, p_y)   3 -> NS residuals
template <int NST, int MODE>
__global__ __launch_bounds__(32) void pinn_fwd(
    const float* __restrict__ xy, int npts,
    const float* __restrict__ W1, const float* __restrict__ b1,
    const __bf16* __restrict__ Wpk,
    const float* __restrict__ b2, const float* __restrict__ b3,
    const float* __restrict__ b4,
    const float* __restrict__ W5, const float* __restrict__ b5,
    float* __restrict__ out) {
  __shared__ __align__(16) __bf16 act[NST * TILE_M * HIDDEN];
  __shared__ float sxy[TILE_M * 2];
  int lane = threadIdx.x;
  int tile = blockIdx.x;
  (void)npts;

  sxy[lane] = xy[tile * (TILE_M * 2) + lane];
  __syncthreads();  // single wave: DS in-order, barrier lowers to S_NOP

  // ---- layer 1: 2 -> 128 + tanh, scalar; seed all derivative states ----
  for (int i = lane; i < TILE_M * HIDDEN; i += 32) {
    int r = i >> 7, c = i & 127;
    float x = sxy[r * 2 + 0], y = sxy[r * 2 + 1];
    float w0 = W1[c], w1 = W1[HIDDEN + c];
    float t = tanhf(x * w0 + y * w1 + b1[c]);
    act[0 * 2048 + r * 128 + c] = (__bf16)t;
    if (NST >= 3) {
      float tp = 1.f - t * t;
      act[1 * 2048 + r * 128 + c] = (__bf16)(tp * w0);
      act[2 * 2048 + r * 128 + c] = (__bf16)(tp * w1);
      if (NST == 5) {
        float tpp = -2.f * t * tp;
        act[3 * 2048 + r * 128 + c] = (__bf16)(tpp * w0 * w0);
        act[4 * 2048 + r * 128 + c] = (__bf16)(tpp * w1 * w1);
      }
    }
  }
  __syncthreads();

  // D-fragment coordinates for this lane (f32 16x16 C/D layout)
  int col0  = lane & 15;
  int rbase = (lane >= 16) ? 8 : 0;
  int arow  = lane & 15;
  int aoff  = (lane >= 16) ? 16 : 0;

  // ---- hidden layers 2..4 via WMMA; all states in flight per N-tile ----
  for (int L = 0; L < 3; L++) {
    const __bf16* wl = Wpk + L * (4 * 8 * 32 * 16);
    const float* bL  = (L == 0) ? b2 : ((L == 1) ? b3 : b4);
    if (L < 2)  // pull next layer's packed weights toward the WGP
      __builtin_prefetch(Wpk + (L + 1) * (4 * 8 * 32 * 16), 0, 1);

    // A fragments for every state (read before any store of this layer)
    AFrag a[NST][4];
#pragma unroll
    for (int s = 0; s < NST; s++)
#pragma unroll
      for (int kb = 0; kb < 4; kb++) {
        const __bf16* p = act + s * 2048 + arow * HIDDEN + kb * 32 + aoff;
        a[s][kb].q[0] = *(const uint4*)(p);
        a[s][kb].q[1] = *(const uint4*)(p + 8);
      }

#pragma unroll
    for (int nt = 0; nt < 8; nt++) {
      // one B-fragment set per N-tile, reused across all NST states
      bf16x16 b[4];
#pragma unroll
      for (int kb = 0; kb < 4; kb++)
        b[kb] = *(const bf16x16*)(wl + ((kb * 8 + nt) * 32 + lane) * 16);

      f32x8 c[NST];
#pragma unroll
      for (int s = 0; s < NST; s++)
        c[s] = (f32x8){0.f, 0.f, 0.f, 0.f, 0.f, 0.f, 0.f, 0.f};
#pragma unroll
      for (int kb = 0; kb < 4; kb++)
#pragma unroll
        for (int s = 0; s < NST; s++)
          c[s] = __builtin_amdgcn_wmma_f32_16x16x32_bf16(
              false, a[s][kb].v, false, b[kb], (short)0, c[s], false, false);

      // tanh chain rule, elementwise on the co-resident D fragments
      float bc = bL[nt * 16 + col0];
#pragma unroll
      for (int g = 0; g < 8; g++) {
        int ofs = (rbase + g) * 128 + nt * 16 + col0;
        float t = tanhf(c[0][g] + bc);
        act[0 * 2048 + ofs] = (__bf16)t;
        if (NST >= 3) {
          float tp = 1.f - t * t;
          float zx = c[1][g], zy = c[2][g];
          act[1 * 2048 + ofs] = (__bf16)(tp * zx);
          act[2 * 2048 + ofs] = (__bf16)(tp * zy);
          if (NST == 5) {
            float tpp = -2.f * t * tp;
            act[3 * 2048 + ofs] = (__bf16)(tpp * zx * zx + tp * c[3][g]);
            act[4 * 2048 + ofs] = (__bf16)(tpp * zy * zy + tp * c[4][g]);
          }
        }
      }
    }
  }
  __syncthreads();

  // ---- output layer (128 -> 3, linear): all 32 lanes, split-K + shuffle ----
  {
    int row   = lane & 15;
    int kbase = (lane >> 4) << 6;  // 0 or 64
    float V0 = 0, V1 = 0, V2 = 0;
    float Jx0 = 0, Jx1 = 0, Jx2 = 0, Jy0 = 0, Jy1 = 0, Jy2 = 0;
    float Hx0 = 0, Hx1 = 0, Hy0 = 0, Hy1 = 0;
    for (int k = kbase; k < kbase + 64; k++) {
      float w0 = W5[k * 3 + 0], w1 = W5[k * 3 + 1], w2 = W5[k * 3 + 2];
      float h = (float)act[0 * 2048 + row * 128 + k];
      V0 += h * w0; V1 += h * w1; V2 += h * w2;
      if (NST >= 3) {
        float hx = (float)act[1 * 2048 + row * 128 + k];
        float hy = (float)act[2 * 2048 + row * 128 + k];
        Jx0 += hx * w0; Jx1 += hx * w1; Jx2 += hx * w2;
        Jy0 += hy * w0; Jy1 += hy * w1; Jy2 += hy * w2;
        if (NST == 5) {
          float hxx = (float)act[3 * 2048 + row * 128 + k];
          float hyy = (float)act[4 * 2048 + row * 128 + k];
          Hx0 += hxx * w0; Hx1 += hxx * w1;
          Hy0 += hyy * w0; Hy1 += hyy * w1;
        }
      }
    }
    // cross-half reduction (lane ^ 16)
    V0 += __shfl_xor(V0, 16, 32);
    V1 += __shfl_xor(V1, 16, 32);
    V2 += __shfl_xor(V2, 16, 32);
    if (NST >= 3) {
      Jx0 += __shfl_xor(Jx0, 16, 32);
      Jx1 += __shfl_xor(Jx1, 16, 32);
      Jx2 += __shfl_xor(Jx2, 16, 32);
      Jy0 += __shfl_xor(Jy0, 16, 32);
      Jy1 += __shfl_xor(Jy1, 16, 32);
      Jy2 += __shfl_xor(Jy2, 16, 32);
      if (NST == 5) {
        Hx0 += __shfl_xor(Hx0, 16, 32);
        Hx1 += __shfl_xor(Hx1, 16, 32);
        Hy0 += __shfl_xor(Hy0, 16, 32);
        Hy1 += __shfl_xor(Hy1, 16, 32);
      }
    }
    if (lane < 16) {
      V0 += b5[0]; V1 += b5[1]; V2 += b5[2];
      float o0, o1, o2;
      if (MODE == 0)      { o0 = V0; o1 = V1; o2 = V0; }
      else if (MODE == 1) { o0 = V2; o1 = V2; o2 = V2; }
      else if (MODE == 2) { o0 = V0; o1 = V1; o2 = Jy2; }
      else {
        // RHO = 1, MU = 0.01
        o0 = V0 * Jx0 + V1 * Jy0 + Jx2 - 0.01f * (Hx0 + Hy0);
        o1 = V0 * Jx1 + V1 * Jy1 + Jy2 - 0.01f * (Hx1 + Hy1);
        o2 = Jx0 + Jy1;
      }
      int gp = tile * TILE_M + row;
      out[gp * 3 + 0] = o0;
      out[gp * 3 + 1] = o1;
      out[gp * 3 + 2] = o2;
    }
  }
}

extern "C" void kernel_launch(void* const* d_in, const int* in_sizes, int n_in,
                              void* d_out, int out_size, void* d_ws, size_t ws_size,
                              hipStream_t stream) {
  (void)n_in; (void)out_size; (void)ws_size;
  const float* xy_eqn    = (const float*)d_in[0];
  const float* xy_in     = (const float*)d_in[1];
  const float* xy_out    = (const float*)d_in[2];
  const float* xy_w1     = (const float*)d_in[3];
  const float* xy_w2     = (const float*)d_in[4];
  const float* xy_circle = (const float*)d_in[5];
  const float* W1 = (const float*)d_in[6];
  const float* b1 = (const float*)d_in[7];
  const float* W2 = (const float*)d_in[8];
  const float* b2 = (const float*)d_in[9];
  const float* W3 = (const float*)d_in[10];
  const float* b3 = (const float*)d_in[11];
  const float* W4 = (const float*)d_in[12];
  const float* b4 = (const float*)d_in[13];
  const float* W5 = (const float*)d_in[14];
  const float* b5 = (const float*)d_in[15];

  __bf16* wpk = (__bf16*)d_ws;  // 3*4*8*32*16 bf16 = 96 KB packed weights

  pack_weights_kernel<<<(3 * 4 * 8 * 32 * 16 + 255) / 256, 256, 0, stream>>>(
      W2, W3, W4, wpk);

  int n_eqn = in_sizes[0] / 2;
  int n_i   = in_sizes[1] / 2;
  int n_o   = in_sizes[2] / 2;
  int n_w1  = in_sizes[3] / 2;
  int n_w2  = in_sizes[4] / 2;
  int n_c   = in_sizes[5] / 2;

  float* out = (float*)d_out;
  size_t off = 0;
  pinn_fwd<5, 3><<<n_eqn / 16, 32, 0, stream>>>(xy_eqn, n_eqn, W1, b1, wpk,
                                                b2, b3, b4, W5, b5, out + off);
  off += (size_t)n_eqn * 3;
  pinn_fwd<1, 0><<<n_i / 16, 32, 0, stream>>>(xy_in, n_i, W1, b1, wpk,
                                              b2, b3, b4, W5, b5, out + off);
  off += (size_t)n_i * 3;
  pinn_fwd<1, 1><<<n_o / 16, 32, 0, stream>>>(xy_out, n_o, W1, b1, wpk,
                                              b2, b3, b4, W5, b5, out + off);
  off += (size_t)n_o * 3;
  pinn_fwd<3, 2><<<n_w1 / 16, 32, 0, stream>>>(xy_w1, n_w1, W1, b1, wpk,
                                               b2, b3, b4, W5, b5, out + off);
  off += (size_t)n_w1 * 3;
  pinn_fwd<3, 2><<<n_w2 / 16, 32, 0, stream>>>(xy_w2, n_w2, W1, b1, wpk,
                                               b2, b3, b4, W5, b5, out + off);
  off += (size_t)n_w2 * 3;
  pinn_fwd<1, 0><<<n_c / 16, 32, 0, stream>>>(xy_circle, n_c, W1, b1, wpk,
                                              b2, b3, b4, W5, b5, out + off);
}
